// Mamba2Block_6786048327772
// MI455X (gfx1250) — matrix operand order repaired and checked by reading, not compiled, not run on hardware
//
#include <hip/hip_runtime.h>
#include <hip/hip_bf16.h>
#include <stdint.h>

// ---------------------------------------------------------------------------
// Types for CDNA5 WMMA (wave32)
// ---------------------------------------------------------------------------
typedef __bf16 bf16;
typedef __attribute__((ext_vector_type(16))) __bf16 v16bf;
typedef __attribute__((ext_vector_type(8)))  float  v8f;

#define D_MODEL   1024
#define D_STATE   128
#define N_HEADS   32
#define CHUNK     64
#define D_INNER   2048
#define CONV_DIM  2304
#define D_IN_PROJ 4384
#define HEAD_DIM  64
#define BATCH     4
#define SEQ       4096
#define ROWS      (BATCH * SEQ)     // 16384
#define NCHUNK    (SEQ / CHUNK)     // 64

// ---------------------------------------------------------------------------
// bf16 helpers (round-to-nearest-even)
// ---------------------------------------------------------------------------
__device__ __forceinline__ bf16 f2bf(float x) {
  unsigned u = __builtin_bit_cast(unsigned, x);
  u += 0x7FFFu + ((u >> 16) & 1u);
  unsigned short h = (unsigned short)(u >> 16);
  return __builtin_bit_cast(bf16, h);
}
__device__ __forceinline__ float bf2f(bf16 b) {
  unsigned short h = __builtin_bit_cast(unsigned short, b);
  unsigned u = ((unsigned)h) << 16;
  return __builtin_bit_cast(float, u);
}
__device__ __forceinline__ v8f vzero8() {
  v8f z;
#pragma unroll
  for (int i = 0; i < 8; i++) z[i] = 0.f;
  return z;
}

// ---------------------------------------------------------------------------
// WMMA fragment loaders (contiguous-k order, same in A and B; wave32)
// A 16x32 bf16: lane m=lane&15; elems 0..15 -> K = half*16 + i  (K-contiguous per lane)
// B 32x16 bf16: lane n=lane&15; elems 0..15 -> K = half*16 + i  (K-contiguous per lane)
// src points at tile origin (row 0 of the tile, k=0 of the matrix), ld in elements
// ---------------------------------------------------------------------------
__device__ __forceinline__ v16bf load_fragA(const bf16* __restrict__ src, int ld, int k0) {
  int lane = threadIdx.x & 31;
  int half = lane >> 4;
  const bf16* q = src + (size_t)(lane & 15) * ld + k0 + half * 16;
  v16bf f;
#pragma unroll
  for (int i = 0; i < 16; i++) f[i] = q[i];
  return f;
}
__device__ __forceinline__ v16bf load_fragB(const bf16* __restrict__ src, int ld, int k0) {
  int lane = threadIdx.x & 31;
  int half = lane >> 4;
  const bf16* q = src + (size_t)(lane & 15) * ld + k0 + half * 16;
  v16bf f;
#pragma unroll
  for (int i = 0; i < 16; i++) f[i] = q[i];
  return f;
}
__device__ __forceinline__ v16bf load_fragB_guard(const bf16* __restrict__ src, int ld,
                                                  int k0, int cols_valid) {
  int lane = threadIdx.x & 31;
  int half = lane >> 4;
  int col  = lane & 15;
  v16bf f;
  if (col < cols_valid) {
    const bf16* q = src + (size_t)col * ld + k0 + half * 16;
#pragma unroll
    for (int i = 0; i < 16; i++) f[i] = q[i];
  } else {
#pragma unroll
    for (int i = 0; i < 16; i++) f[i] = __builtin_bit_cast(bf16, (unsigned short)0);
  }
  return f;
}
// B-frag with per-K scaling (scale indexed by matrix-local k)
__device__ __forceinline__ v16bf load_fragB_scaled(const bf16* __restrict__ src, int ld,
                                                   int k0, const float* __restrict__ scale) {
  int lane = threadIdx.x & 31;
  int half = lane >> 4;
  const bf16* q = src + (size_t)(lane & 15) * ld + k0 + half * 16;
  v16bf f;
#pragma unroll
  for (int i = 0; i < 16; i++) f[i] = f2bf(bf2f(q[i]) * scale[k0 + half * 16 + i]);
  return f;
}
// B-frag from fp32 (LDS-resident state matrix)
__device__ __forceinline__ v16bf load_fragB_f32(const float* __restrict__ src, int ld, int k0) {
  int lane = threadIdx.x & 31;
  int half = lane >> 4;
  const float* q = src + (size_t)(lane & 15) * ld + k0 + half * 16;
  v16bf f;
#pragma unroll
  for (int i = 0; i < 16; i++) f[i] = f2bf(q[i]);
  return f;
}
__device__ __forceinline__ v8f wmma_bf16(v16bf a, v16bf b, v8f c) {
  return __builtin_amdgcn_wmma_f32_16x16x32_bf16(false, a, false, b, (short)0, c, false, false);
}

// ---------------------------------------------------------------------------
// Generic NT WMMA GEMM:  C(MxN) = A(MxK) @ B(NxK)^T, bf16 in, fp32 out.
// Block: 128 threads = 4 waves; block tile 128x128, wave tile 64x64.
// EPI==0 : plain row-major store to C0 (ldc = N)
// EPI==1 : split store for zxbcdt -> C0 = z (2048), C1 = xBC (2304), C2 = dt (32)
// ---------------------------------------------------------------------------
template <int EPI>
__global__ __launch_bounds__(128) void wmma_gemm_nt(const bf16* __restrict__ A,
                                                    const bf16* __restrict__ B,
                                                    float* __restrict__ C0,
                                                    float* __restrict__ C1,
                                                    float* __restrict__ C2,
                                                    int M, int N, int K) {
  const int lane = threadIdx.x & 31;
  const int wave = threadIdx.x >> 5;
  const int bm = blockIdx.y * 128 + (wave >> 1) * 64;
  const int bn = blockIdx.x * 128 + (wave & 1) * 64;

  v8f acc[4][4];
#pragma unroll
  for (int i = 0; i < 4; i++)
#pragma unroll
    for (int j = 0; j < 4; j++) acc[i][j] = vzero8();

  for (int k0 = 0; k0 < K; k0 += 32) {
    v16bf af[4], bfg[4];
#pragma unroll
    for (int i = 0; i < 4; i++)
      af[i] = load_fragA(A + (size_t)(bm + i * 16) * K, K, k0);
#pragma unroll
    for (int j = 0; j < 4; j++)
      bfg[j] = load_fragB_guard(B + (size_t)(bn + j * 16) * K, K, k0, N - (bn + j * 16));
#pragma unroll
    for (int i = 0; i < 4; i++)
#pragma unroll
      for (int j = 0; j < 4; j++)
        acc[i][j] = wmma_bf16(af[i], bfg[j], acc[i][j]);
  }

  const int half = lane >> 4, nl = lane & 15;
#pragma unroll
  for (int i = 0; i < 4; i++)
#pragma unroll
    for (int j = 0; j < 4; j++)
#pragma unroll
      for (int v = 0; v < 8; v++) {
        int m = bm + i * 16 + half * 8 + v;
        int n = bn + j * 16 + nl;
        float val = acc[i][j][v];
        if (EPI == 0) {
          if (n < N) C0[(size_t)m * N + n] = val;
        } else {
          if (n < D_INNER)                      C0[(size_t)m * D_INNER + n] = val;
          else if (n < D_INNER + CONV_DIM)      C1[(size_t)m * CONV_DIM + (n - D_INNER)] = val;
          else if (n < D_IN_PROJ)               C2[(size_t)m * N_HEADS + (n - D_INNER - CONV_DIM)] = val;
        }
      }
}

// ---------------------------------------------------------------------------
// Elementwise kernels
// ---------------------------------------------------------------------------
__global__ void to_bf16_kernel(const float* __restrict__ in, bf16* __restrict__ out, long n) {
  long stride = (long)gridDim.x * blockDim.x;
  for (long i = (long)blockIdx.x * blockDim.x + threadIdx.x; i < n; i += stride)
    out[i] = f2bf(in[i]);
}

// depthwise causal conv (taps 4) + bias + SiLU
__global__ void conv_silu_kernel(const float* __restrict__ xraw, const float* __restrict__ cw,
                                 const float* __restrict__ cb, float* __restrict__ xact,
                                 long total) {
  long i = (long)blockIdx.x * blockDim.x + threadIdx.x;
  if (i >= total) return;
  int ch = (int)(i % CONV_DIM);
  long row = i / CONV_DIM;
  int t = (int)(row % SEQ);
  float acc = cb[ch];
#pragma unroll
  for (int k = 0; k < 4; k++) {
    int tt = t + k - 3;
    if (tt >= 0) acc += cw[ch * 4 + k] * xraw[(row + (long)(k - 3)) * CONV_DIM + ch];
  }
  xact[i] = acc / (1.f + __expf(-acc));
}

// dt softplus + A*dt
__global__ void pack_dt_kernel(const float* __restrict__ dtraw, const float* __restrict__ dt_bias,
                               const float* __restrict__ A_log, float* __restrict__ dt_soft,
                               float* __restrict__ dtA, long total) {
  long i = (long)blockIdx.x * blockDim.x + threadIdx.x;
  if (i >= total) return;
  int h = (int)(i & (N_HEADS - 1));
  float x = dtraw[i] + dt_bias[h];
  float sp = (x > 20.f) ? x : log1pf(__expf(x));
  dt_soft[i] = sp;
  dtA[i] = -__expf(A_log[h]) * sp;
}

// B/C -> bf16 in both (b,t,n) and transposed (b,n,t) layouts
__global__ void pack_bc_kernel(const float* __restrict__ xact, bf16* __restrict__ Bn,
                               bf16* __restrict__ Cn, bf16* __restrict__ Bt,
                               bf16* __restrict__ Ct, long total) {
  long i = (long)blockIdx.x * blockDim.x + threadIdx.x;
  if (i >= total) return;
  int n = (int)(i & (D_STATE - 1));
  long row = i >> 7;
  int b = (int)(row >> 12);
  int t = (int)(row & (SEQ - 1));
  float vb = xact[row * CONV_DIM + D_INNER + n];
  float vc = xact[row * CONV_DIM + D_INNER + D_STATE + n];
  Bn[i] = f2bf(vb);
  Cn[i] = f2bf(vc);
  size_t tr = ((size_t)(b * D_STATE + n)) * SEQ + t;
  Bt[tr] = f2bf(vb);
  Ct[tr] = f2bf(vc);
}

// x * dt -> bf16, layout (b, h, p, t) (t-contiguous)
__global__ void pack_x_kernel(const float* __restrict__ xact, const float* __restrict__ dt_soft,
                              bf16* __restrict__ xdtT, long total) {
  long i = (long)blockIdx.x * blockDim.x + threadIdx.x;
  if (i >= total) return;
  int t = (int)(i & (SEQ - 1));
  long r = i >> 12;
  int p = (int)(r & (HEAD_DIM - 1)); r >>= 6;
  int h = (int)(r & (N_HEADS - 1));
  int b = (int)(r >> 5);
  long row = (long)b * SEQ + t;
  float v = xact[row * CONV_DIM + h * HEAD_DIM + p] * dt_soft[row * N_HEADS + h];
  xdtT[i] = f2bf(v);
}

// ---------------------------------------------------------------------------
// SSD phase A: per (b, chunk, head), 1 wave.
//  G1: CBt = C(64x128) @ B^T           -> mask with L = exp(segsum) -> M (LDS bf16)
//  G2: Y_diag = M(64x64) @ Xdt(64x64)  -> store into Y
//  G3: states(p,n) = Xdt^T @ (decay o B) -> chunk states to global
// Also writes per-chunk inclusive cumsum of A*dt (cumA) for phase B.
// ---------------------------------------------------------------------------
__global__ __launch_bounds__(32) void ssd_chunk_kernel(
    const bf16* __restrict__ Cn, const bf16* __restrict__ Bn, const bf16* __restrict__ Bt,
    const bf16* __restrict__ xdtT, const float* __restrict__ dtA, float* __restrict__ Y,
    float* __restrict__ states, float* __restrict__ cumA) {
  const int c = blockIdx.x, h = blockIdx.y, b = blockIdx.z;
  const int lane = threadIdx.x;
  __shared__ float sCum[CHUNK];
  __shared__ float sScale[CHUNK];
  __shared__ bf16 sM[CHUNK * CHUNK];

  if (lane == 0) {
    float s = 0.f;
    const float* dp = dtA + ((size_t)(b * SEQ + c * CHUNK)) * N_HEADS + h;
    float* cp = cumA + ((size_t)((b * NCHUNK + c) * N_HEADS + h)) * CHUNK;
    for (int t = 0; t < CHUNK; t++) { s += dp[(size_t)t * N_HEADS]; sCum[t] = s; cp[t] = s; }
  }
  __syncthreads();
  {
    float last = sCum[CHUNK - 1];
    for (int t = lane; t < CHUNK; t += 32) sScale[t] = __expf(last - sCum[t]);
  }
  __syncthreads();

  const bf16* Cb = Cn + ((size_t)(b * SEQ + c * CHUNK)) * D_STATE;
  const bf16* Bb = Bn + ((size_t)(b * SEQ + c * CHUNK)) * D_STATE;
  const bf16* Xb = xdtT + ((size_t)((b * N_HEADS + h) * HEAD_DIM)) * SEQ + c * CHUNK;
  const int half = lane >> 4, nl = lane & 15;

  // ---- G1: CBt (64x64, K=128) ----
  {
    v8f acc[4][4];
#pragma unroll
    for (int i = 0; i < 4; i++)
#pragma unroll
      for (int j = 0; j < 4; j++) acc[i][j] = vzero8();
#pragma unroll
    for (int kk = 0; kk < 4; kk++) {
      int k0 = kk * 32;
      v16bf af[4], bfg[4];
#pragma unroll
      for (int i = 0; i < 4; i++) af[i] = load_fragA(Cb + (size_t)(i * 16) * D_STATE, D_STATE, k0);
#pragma unroll
      for (int j = 0; j < 4; j++) bfg[j] = load_fragB(Bb + (size_t)(j * 16) * D_STATE, D_STATE, k0);
#pragma unroll
      for (int i = 0; i < 4; i++)
#pragma unroll
        for (int j = 0; j < 4; j++) acc[i][j] = wmma_bf16(af[i], bfg[j], acc[i][j]);
    }
    // mask with 1-semiseparable L and stage M into LDS
#pragma unroll
    for (int i = 0; i < 4; i++)
#pragma unroll
      for (int j = 0; j < 4; j++)
#pragma unroll
        for (int v = 0; v < 8; v++) {
          int t = i * 16 + half * 8 + v;
          int s_ = j * 16 + nl;
          float val = (s_ <= t) ? acc[i][j][v] * __expf(sCum[t] - sCum[s_]) : 0.f;
          sM[t * CHUNK + s_] = f2bf(val);
        }
  }
  __syncthreads();

  // ---- G2: Y_diag = M @ Xdt (64x64, K=64) ----
  {
    v8f acc[4][4];
#pragma unroll
    for (int i = 0; i < 4; i++)
#pragma unroll
      for (int j = 0; j < 4; j++) acc[i][j] = vzero8();
#pragma unroll
    for (int kk = 0; kk < 2; kk++) {
      int k0 = kk * 32;
      v16bf af[4], bfg[4];
#pragma unroll
      for (int i = 0; i < 4; i++) af[i] = load_fragA(sM + (size_t)(i * 16) * CHUNK, CHUNK, k0);
#pragma unroll
      for (int j = 0; j < 4; j++) bfg[j] = load_fragB(Xb + (size_t)(j * 16) * SEQ, SEQ, k0);
#pragma unroll
      for (int i = 0; i < 4; i++)
#pragma unroll
        for (int j = 0; j < 4; j++) acc[i][j] = wmma_bf16(af[i], bfg[j], acc[i][j]);
    }
    float* Yb = Y + ((size_t)(b * SEQ + c * CHUNK)) * D_INNER + h * HEAD_DIM;
#pragma unroll
    for (int i = 0; i < 4; i++)
#pragma unroll
      for (int j = 0; j < 4; j++)
#pragma unroll
        for (int v = 0; v < 8; v++) {
          int t = i * 16 + half * 8 + v;
          int p = j * 16 + nl;
          Yb[(size_t)t * D_INNER + p] = acc[i][j][v];
        }
  }

  // ---- G3: chunk states (64p x 128n, K=64t), decay folded into B-frag ----
  float* Sb = states + ((size_t)(((b * NCHUNK + c) * N_HEADS + h)) * HEAD_DIM) * D_STATE;
#pragma unroll
  for (int nh = 0; nh < 2; nh++) {
    v8f acc[4][4];
#pragma unroll
    for (int i = 0; i < 4; i++)
#pragma unroll
      for (int j = 0; j < 4; j++) acc[i][j] = vzero8();
#pragma unroll
    for (int kk = 0; kk < 2; kk++) {
      int k0 = kk * 32;
      v16bf af[4], bfg[4];
#pragma unroll
      for (int i = 0; i < 4; i++) af[i] = load_fragA(Xb + (size_t)(i * 16) * SEQ, SEQ, k0);
#pragma unroll
      for (int j = 0; j < 4; j++)
        bfg[j] = load_fragB_scaled(Bt + ((size_t)(b * D_STATE + nh * 64 + j * 16)) * SEQ + c * CHUNK,
                                   SEQ, k0, sScale);
#pragma unroll
      for (int i = 0; i < 4; i++)
#pragma unroll
        for (int j = 0; j < 4; j++) acc[i][j] = wmma_bf16(af[i], bfg[j], acc[i][j]);
    }
#pragma unroll
    for (int i = 0; i < 4; i++)
#pragma unroll
      for (int j = 0; j < 4; j++)
#pragma unroll
        for (int v = 0; v < 8; v++) {
          int p = i * 16 + half * 8 + v;
          int n = nh * 64 + j * 16 + nl;
          Sb[(size_t)p * D_STATE + n] = acc[i][j][v];
        }
  }
}

// ---------------------------------------------------------------------------
// SSD phase B: per (b, head), 4 waves. Sequential over 64 chunks.
// S (64p x 128n) lives in LDS (fp32). Per chunk:
//   Y_off = (C @ S^T) * exp(cumA[t])  added into Y;  S = S*exp(cumA[63]) + states_c
// ---------------------------------------------------------------------------
__global__ __launch_bounds__(128) void ssd_scan_kernel(const bf16* __restrict__ Cn,
                                                       const float* __restrict__ states,
                                                       const float* __restrict__ cumA,
                                                       float* __restrict__ Y) {
  const int h = blockIdx.x, b = blockIdx.y;
  const int tid = threadIdx.x;
  const int lane = tid & 31, wave = tid >> 5;
  const int half = lane >> 4, nl = lane & 15;
  __shared__ float S[HEAD_DIM * D_STATE];   // 32 KB
  __shared__ float sCum[CHUNK];

  for (int e = tid; e < HEAD_DIM * D_STATE; e += 128) S[e] = 0.f;
  __syncthreads();

  for (int c = 0; c < NCHUNK; c++) {
    if (tid < CHUNK)
      sCum[tid] = cumA[((size_t)((b * NCHUNK + c) * N_HEADS + h)) * CHUNK + tid];
    __syncthreads();

    const bf16* Cb = Cn + ((size_t)(b * SEQ + c * CHUNK)) * D_STATE;
    v8f acc[4];
#pragma unroll
    for (int j = 0; j < 4; j++) acc[j] = vzero8();
#pragma unroll
    for (int kk = 0; kk < 4; kk++) {
      int k0 = kk * 32;
      v16bf a = load_fragA(Cb + (size_t)(wave * 16) * D_STATE, D_STATE, k0);
#pragma unroll
      for (int j = 0; j < 4; j++) {
        v16bf bb = load_fragB_f32(S + (size_t)(j * 16) * D_STATE, D_STATE, k0);
        acc[j] = wmma_bf16(a, bb, acc[j]);
      }
    }
    float* Yb = Y + ((size_t)(b * SEQ + c * CHUNK)) * D_INNER + h * HEAD_DIM;
#pragma unroll
    for (int j = 0; j < 4; j++)
#pragma unroll
      for (int v = 0; v < 8; v++) {
        int t = wave * 16 + half * 8 + v;
        int p = j * 16 + nl;
        float sc = __expf(sCum[t]);
        Yb[(size_t)t * D_INNER + p] += acc[j][v] * sc;
      }
    __syncthreads();   // all waves done reading S before update

    float dT = __expf(sCum[CHUNK - 1]);
    const float* stc = states + ((size_t)(((b * NCHUNK + c) * N_HEADS + h)) * HEAD_DIM) * D_STATE;
    for (int e = tid; e < HEAD_DIM * D_STATE; e += 128) S[e] = S[e] * dT + stc[e];
    __syncthreads();
  }
}

// ---------------------------------------------------------------------------
// Gated RMSNorm: t = (Y + x*D[h]) * silu(z); t *= rsqrt(mean(t^2)+eps) * norm_w
// One block per (b, l) row; output bf16 for final GEMM.
// ---------------------------------------------------------------------------
__global__ __launch_bounds__(256) void gate_norm_kernel(const float* __restrict__ Y,
                                                        const float* __restrict__ xact,
                                                        const float* __restrict__ z,
                                                        const float* __restrict__ Dv,
                                                        const float* __restrict__ norm_w,
                                                        bf16* __restrict__ tout) {
  const int row = blockIdx.x;
  const int tid = threadIdx.x;
  __shared__ float red[256];
  float vals[8];
  float ss = 0.f;
#pragma unroll
  for (int i = 0; i < 8; i++) {
    int cidx = tid + i * 256;
    float x = xact[(size_t)row * CONV_DIM + cidx];
    float y = Y[(size_t)row * D_INNER + cidx] + x * Dv[cidx >> 6];
    float zz = z[(size_t)row * D_INNER + cidx];
    float g = zz / (1.f + __expf(-zz));
    float t = y * g;
    vals[i] = t;
    ss += t * t;
  }
  red[tid] = ss;
  __syncthreads();
  for (int s = 128; s > 0; s >>= 1) {
    if (tid < s) red[tid] += red[tid + s];
    __syncthreads();
  }
  float scale = rsqrtf(red[0] / (float)D_INNER + 1e-5f);
#pragma unroll
  for (int i = 0; i < 8; i++) {
    int cidx = tid + i * 256;
    tout[(size_t)row * D_INNER + cidx] = f2bf(vals[i] * scale * norm_w[cidx]);
  }
}

// ---------------------------------------------------------------------------
// Host orchestration
// ---------------------------------------------------------------------------
extern "C" void kernel_launch(void* const* d_in, const int* in_sizes, int n_in,
                              void* d_out, int out_size, void* d_ws, size_t ws_size,
                              hipStream_t stream) {
  (void)in_sizes; (void)n_in; (void)out_size; (void)ws_size;
  const float* u          = (const float*)d_in[0];
  const float* in_proj_w  = (const float*)d_in[1];
  const float* conv_w     = (const float*)d_in[2];
  const float* conv_b     = (const float*)d_in[3];
  const float* dt_bias    = (const float*)d_in[4];
  const float* A_log      = (const float*)d_in[5];
  const float* Dv         = (const float*)d_in[6];
  const float* norm_w     = (const float*)d_in[7];
  const float* out_proj_w = (const float*)d_in[8];
  float* out = (float*)d_out;

  char* wsb = (char*)d_ws;
  size_t off = 0;
  auto alloc = [&](size_t bytes) -> char* {
    char* p = wsb + off;
    off += (bytes + 255) & ~(size_t)255;
    return p;
  };
  bf16*  u_bf     = (bf16*) alloc((size_t)ROWS * D_MODEL * 2);
  bf16*  win_bf   = (bf16*) alloc((size_t)D_IN_PROJ * D_MODEL * 2);
  bf16*  wout_bf  = (bf16*) alloc((size_t)D_MODEL * D_INNER * 2);
  float* z_f      = (float*)alloc((size_t)ROWS * D_INNER * 4);
  float* xBC_raw  = (float*)alloc((size_t)ROWS * CONV_DIM * 4);
  float* dt_raw   = (float*)alloc((size_t)ROWS * N_HEADS * 4);
  float* xBC_act  = (float*)alloc((size_t)ROWS * CONV_DIM * 4);
  float* dt_soft  = (float*)alloc((size_t)ROWS * N_HEADS * 4);
  float* dtA      = (float*)alloc((size_t)ROWS * N_HEADS * 4);
  bf16*  xdtT     = (bf16*) alloc((size_t)ROWS * D_INNER * 2);
  bf16*  Bn       = (bf16*) alloc((size_t)ROWS * D_STATE * 2);
  bf16*  Cn       = (bf16*) alloc((size_t)ROWS * D_STATE * 2);
  bf16*  Bt       = (bf16*) alloc((size_t)ROWS * D_STATE * 2);
  bf16*  Ct       = (bf16*) alloc((size_t)ROWS * D_STATE * 2);
  float* Ybuf     = (float*)alloc((size_t)ROWS * D_INNER * 4);
  float* states   = (float*)alloc((size_t)BATCH * NCHUNK * N_HEADS * HEAD_DIM * D_STATE * 4);
  float* cumA     = (float*)alloc((size_t)BATCH * NCHUNK * N_HEADS * CHUNK * 4);
  bf16*  t_bf     = (bf16*) alloc((size_t)ROWS * D_INNER * 2);
  (void)Ct;

  // 1. fp32 -> bf16 conversions
  {
    long n1 = (long)ROWS * D_MODEL;
    to_bf16_kernel<<<8192, 256, 0, stream>>>(u, u_bf, n1);
    long n2 = (long)D_IN_PROJ * D_MODEL;
    to_bf16_kernel<<<8192, 256, 0, stream>>>(in_proj_w, win_bf, n2);
    long n3 = (long)D_MODEL * D_INNER;
    to_bf16_kernel<<<8192, 256, 0, stream>>>(out_proj_w, wout_bf, n3);
  }
  // 2. in_proj GEMM: zxbcdt = u @ W^T (split epilogue)
  wmma_gemm_nt<1><<<dim3((D_IN_PROJ + 127) / 128, ROWS / 128), 128, 0, stream>>>(
      u_bf, win_bf, z_f, xBC_raw, dt_raw, ROWS, D_IN_PROJ, D_MODEL);
  // 3. depthwise causal conv + SiLU
  {
    long total = (long)ROWS * CONV_DIM;
    conv_silu_kernel<<<(int)((total + 255) / 256), 256, 0, stream>>>(xBC_raw, conv_w, conv_b,
                                                                     xBC_act, total);
  }
  // 4. packing: dt softplus / A*dt, B/C bf16 (both layouts), x*dt transposed bf16
  {
    long tdt = (long)ROWS * N_HEADS;
    pack_dt_kernel<<<(int)((tdt + 255) / 256), 256, 0, stream>>>(dt_raw, dt_bias, A_log, dt_soft,
                                                                 dtA, tdt);
    long tbc = (long)ROWS * D_STATE;
    pack_bc_kernel<<<(int)((tbc + 255) / 256), 256, 0, stream>>>(xBC_act, Bn, Cn, Bt, Ct, tbc);
    long tx = (long)ROWS * D_INNER;
    pack_x_kernel<<<(int)((tx + 255) / 256), 256, 0, stream>>>(xBC_act, dt_soft, xdtT, tx);
  }
  // 5. SSD phase A (per chunk-head, all parallel)
  ssd_chunk_kernel<<<dim3(NCHUNK, N_HEADS, BATCH), 32, 0, stream>>>(Cn, Bn, Bt, xdtT, dtA, Ybuf,
                                                                    states, cumA);
  // 6. SSD phase B (sequential inter-chunk scan per (b,h))
  ssd_scan_kernel<<<dim3(N_HEADS, BATCH), 128, 0, stream>>>(Cn, states, cumA, Ybuf);
  // 7. gated RMSNorm -> bf16
  gate_norm_kernel<<<ROWS, 256, 0, stream>>>(Ybuf, xBC_act, z_f, Dv, norm_w, t_bf);
  // 8. out_proj GEMM -> d_out
  wmma_gemm_nt<0><<<dim3(D_MODEL / 128, ROWS / 128), 128, 0, stream>>>(
      t_bf, wout_bf, out, nullptr, nullptr, ROWS, D_MODEL, D_INNER);
}
